// MultiHeadAttention_33827162423385
// MI455X (gfx1250) — matrix operand order repaired and checked by reading, not compiled, not run on hardware
//
#include <hip/hip_runtime.h>
#include <hip/hip_bf16.h>
#include <math.h>

// ---------------------------------------------------------------------------
// MultiHeadAttention (B=4, S=1024, D=768, H=12, full-width heads) for gfx1250.
// All matrix math on v_wmma_f32_16x16x32_bf16 (f32 accumulation).
// - A tiles stream into LDS via the Tensor Data Mover (tensor_load_to_lds,
//   TENSORcnt), TDM row padding gives the bank-conflict-free 72-bf16 pitch.
// - Double-buffered LDS: next tile's TDM + B copies overlap current WMMAs.
// - All B operands are pre-transposed once ([N][K] bf16) so every B stage is
//   16-byte vector copies; V projection writes v^T directly.
// ---------------------------------------------------------------------------

typedef __bf16 v16bf __attribute__((ext_vector_type(16)));
typedef float  v8f   __attribute__((ext_vector_type(8)));
typedef int    v4i   __attribute__((ext_vector_type(4)));
typedef int    v8i   __attribute__((ext_vector_type(8)));

__global__ void mha_cvt_f32_bf16(const float* __restrict__ src,
                                 __bf16* __restrict__ dst, long long n) {
  long long i = (long long)blockIdx.x * blockDim.x + threadIdx.x;
  long long stride = (long long)gridDim.x * blockDim.x;
  for (; i < n; i += stride) dst[i] = (__bf16)src[i];
}

// Tiled transpose + f32->bf16: dst[z][c][r] = src[z][r][c]; rows,cols % 32 == 0.
__global__ __launch_bounds__(256)
void mha_cvt_t_f32_bf16(const float* __restrict__ src, __bf16* __restrict__ dst,
                        int rows, int cols, long long sZ, long long dZ) {
  __shared__ __bf16 t[32][33];
  const int r0 = blockIdx.x * 32, c0 = blockIdx.y * 32;
  src += (long long)blockIdx.z * sZ;
  dst += (long long)blockIdx.z * dZ;
  const int tx = threadIdx.x & 31, ty = threadIdx.x >> 5;
  for (int j = ty; j < 32; j += 8)
    t[j][tx] = (__bf16)src[(long long)(r0 + j) * cols + (c0 + tx)];
  __syncthreads();
  for (int j = ty; j < 32; j += 8)
    dst[(long long)(c0 + j) * rows + (r0 + tx)] = t[tx][j];
}

// bf16 WMMA GEMM: C[z] = alpha*(A[z] @ B[z]) (+bias)(+causal mask).
// A: [M][K] row-major.  B: given TRANSPOSED, memory layout [N][K] (ldb = row
// stride of B^T).  128x128 block tile, K-step 64, double-buffered LDS.
// z = blockIdx.z decomposed as b = z / Hdiv, h = z % Hdiv for the strides.
// outT: store C transposed (C[n][m], ldc = row stride); outF32: f32 output.
__global__ __launch_bounds__(256, 1)
void mha_gemm_wmma_bf16(const __bf16* __restrict__ A,
                        const __bf16* __restrict__ Bt,
                        void* __restrict__ C,
                        const float* __restrict__ bias,
                        int K, int lda, int ldb, int ldc, int Hdiv,
                        long long aSB, long long aSH,
                        long long bSB, long long bSH,
                        long long cSB, long long cSH,
                        long long biasSH,
                        float alpha, int causal, int outT, int outF32) {
  constexpr int TM = 128, TN = 128, TK = 64, LDSP = 72;  // 144B row pitch
  __shared__ __align__(16) __bf16 As[2][TM][LDSP];       // filled by TDM
  __shared__ __align__(16) __bf16 Bs[2][TN][LDSP];       // B^T tiles: [n][k]

  const int z = blockIdx.z;
  const int h = z % Hdiv;
  const int b = z / Hdiv;
  const __bf16* Ag = A + (long long)b * aSB + (long long)h * aSH;
  const __bf16* Bg = Bt + (long long)b * bSB + (long long)h * bSH;
  const float*  biasp = bias ? bias + (long long)h * biasSH : nullptr;

  const int mBase = blockIdx.x * TM;
  const int nBase = blockIdx.y * TN;

  const int tid  = threadIdx.x;
  const int lane = tid & 31;
  const int half = lane >> 4;         // lane group 0..15 / 16..31
  const int l16  = lane & 15;
  const int wid  = tid >> 5;          // 8 waves
  const int waveM = (wid >> 1) * 32;  // 4 wave-rows of 32
  const int waveN = (wid & 1) * 64;   // 2 wave-cols of 64

  // LDS byte offset (generic LDS VA: low 32 bits are the LDS address).
  const unsigned ldsA = (unsigned)(uintptr_t)(void*)&As[0][0][0];
  constexpr unsigned aBufBytes = TM * LDSP * 2;

  // Issue TDM load of A tile kt into LDS buffer buf (wave 0 only).
  // D# per CDNA5 ISA: data_size=2B; pad every 32 DWORDs
  // (one 64-elem row) by 4 DWORDs (8 bf16) -> LDS pitch 72 bf16 == LDSP.
  auto issueA = [&](int kt, int buf) {
    if (wid != 0) return;
    unsigned long long ga = (unsigned long long)(uintptr_t)(
        Ag + (long long)mBase * lda + kt * TK);
    v4i g0;
    g0[0] = 1;                                  // count=1 (valid user D#)
    g0[1] = (int)(ldsA + buf * aBufBytes);      // lds_addr
    g0[2] = (int)(ga & 0xFFFFFFFFull);          // global_addr[31:0]
    g0[3] = (int)(((ga >> 32) & 0x01FFFFFFull) | (2u << 30));  // type=2
    v8i g1;
    g1[0] = (1 << 16)      // data_size = 2 bytes
          | (1 << 20)      // pad_enable
          | (4 << 22)      // pad_interval: 32 DWORDs (128B = one row)
          | (3 << 25);     // pad_amount: 4 DWORDs (16B)
    g1[1] = (TK << 16);    // tensor_dim0 = 64 (bits 79:48, low half)
    g1[2] = (TM << 16);    // tensor_dim0 hi = 0; tensor_dim1 = 128 (low)
    g1[3] = (TK << 16);    // tensor_dim1 hi = 0; tile_dim0 = 64
    g1[4] = TM;            // tile_dim1 = 128; tile_dim2 = 0
    g1[5] = lda;           // tensor_dim0_stride[31:0]
    g1[6] = 0;             // stride0 hi16 = 0; stride1 low16 = 0
    g1[7] = 0;
    asm volatile("tensor_load_to_lds %0, %1" :: "s"(g0), "s"(g1) : "memory");
  };

  // Stage B^T tile kt into Bs[buf]: straight 16-byte copies, coalesced in k.
  auto stageB = [&](int kt, int buf) {
    const int k0 = kt * TK;
    for (int i = tid; i < TN * (TK / 8); i += 256) {
      int n = i >> 3, c8 = (i & 7) * 8;
      *reinterpret_cast<uint4*>(&Bs[buf][n][c8]) =
          *reinterpret_cast<const uint4*>(
              &Bg[(long long)(nBase + n) * ldb + (k0 + c8)]);
    }
  };

  v8f acc[2][4];
  for (int mi = 0; mi < 2; ++mi)
    for (int ni = 0; ni < 4; ++ni) acc[mi][ni] = v8f{};

  const int kTiles = K / TK;

  // Prologue: fill buffer 0.
  issueA(0, 0);
  stageB(0, 0);
  if (wid == 0) __builtin_amdgcn_s_wait_tensorcnt(0);
  __syncthreads();

  for (int kt = 0; kt < kTiles; ++kt) {
    const int cur = kt & 1, nxt = cur ^ 1;

    // Overlap: kick off next tile (TDM for A, vector copies for B^T).
    if (kt + 1 < kTiles) {
      issueA(kt + 1, nxt);
      stageB(kt + 1, nxt);
      if (kt + 2 < kTiles)  // warm GL2 for the tile after next
        __builtin_prefetch(
            Bg + (long long)(nBase + (tid >> 1)) * ldb + (kt + 2) * TK, 0, 1);
    }

    // ---- two WMMA K-slices on the current LDS tile -----------------------
    for (int ks = 0; ks < 2; ++ks) {
      const int kb = ks * 32;
      // A (16x32 bf16, MxK): lane l16 = row m; elems 0..7 -> k = half*8+0..7,
      //                                      elems 8..15 -> k = 16+half*8+..
      v16bf afrag[2];
      for (int mi = 0; mi < 2; ++mi) {
        const __bf16* ap = &As[cur][waveM + mi * 16 + l16][kb + half * 8];
        uint4* av = reinterpret_cast<uint4*>(&afrag[mi]);
        av[0] = *reinterpret_cast<const uint4*>(ap);
        av[1] = *reinterpret_cast<const uint4*>(ap + 16);
      }
      // B (32x16 bf16, KxN): lane l16 = col n; elems 0..7 -> k = half*8+0..7,
      //                                      elems 8..15 -> k = 16+half*8+..
      v16bf bfrag[4];
      for (int ni = 0; ni < 4; ++ni) {
        const __bf16* bp = &Bs[cur][waveN + ni * 16 + l16][kb + half * 8];
        uint4* bv = reinterpret_cast<uint4*>(&bfrag[ni]);
        bv[0] = *reinterpret_cast<const uint4*>(bp);
        bv[1] = *reinterpret_cast<const uint4*>(bp + 16);
      }
      for (int mi = 0; mi < 2; ++mi)
        for (int ni = 0; ni < 4; ++ni)
          acc[mi][ni] = __builtin_amdgcn_wmma_f32_16x16x32_bf16(
              false, afrag[mi], false, bfrag[ni], (short)0, acc[mi][ni],
              false, false);
    }

    // One barrier per iteration: next buffers staged, current consumed.
    if (wid == 0) __builtin_amdgcn_s_wait_tensorcnt(0);
    __syncthreads();
  }

  // ---- epilogue (branch-hoisted, incremental addressing) -----------------
  // D layout: n = l16, m = half*8 + r (r = vgpr 0..7).
  const int rowLoc = waveM + half * 8;          // local row of r=0, mi=0
  const int colLoc = waveN + l16;               // local col of ni=0
  const long long czoff = (long long)b * cSB + (long long)h * cSH;

  if (outF32) {                       // final projection: f32, bias, no mask
    float* cp = (float*)C + czoff + (long long)(mBase + rowLoc) * ldc +
                (nBase + colLoc);
    for (int mi = 0; mi < 2; ++mi)
      for (int ni = 0; ni < 4; ++ni) {
        const float bv = biasp ? biasp[colLoc + ni * 16] : 0.0f;
        float* cpp = cp + (long long)(mi * 16) * ldc + ni * 16;
        for (int r = 0; r < 8; ++r)
          cpp[(long long)r * ldc] = acc[mi][ni][r] * alpha + bv;
      }
  } else if (outT) {                  // v^T store: C[n][m], bf16, bias
    __bf16* cp = (__bf16*)C + czoff + (long long)(nBase + colLoc) * ldc +
                 (mBase + rowLoc);
    for (int mi = 0; mi < 2; ++mi)
      for (int ni = 0; ni < 4; ++ni) {
        const float bv = biasp ? biasp[colLoc + ni * 16] : 0.0f;
        __bf16* cpp = cp + (long long)(ni * 16) * ldc + mi * 16;
        for (int r = 0; r < 8; ++r)
          cpp[r] = (__bf16)(acc[mi][ni][r] * alpha + bv);
      }
  } else {                            // normal bf16 (+ optional causal mask)
    __bf16* cp = (__bf16*)C + czoff + (long long)(mBase + rowLoc) * ldc +
                 (nBase + colLoc);
    for (int mi = 0; mi < 2; ++mi)
      for (int ni = 0; ni < 4; ++ni) {
        const int col = nBase + colLoc + ni * 16;
        const float bv = biasp ? biasp[colLoc + ni * 16] : 0.0f;
        const int row0 = mBase + rowLoc + mi * 16;
        __bf16* cpp = cp + (long long)(mi * 16) * ldc + ni * 16;
        for (int r = 0; r < 8; ++r) {
          float val = acc[mi][ni][r] * alpha + bv;
          if (causal && col > row0 + r) val = -1e9f;
          cpp[(long long)r * ldc] = (__bf16)val;
        }
      }
  }
}

// In-place row softmax over bf16 rows of length Scol. One wave per row.
__global__ __launch_bounds__(256)
void mha_softmax_rows(__bf16* __restrict__ P, int Scol) {
  const int wid  = threadIdx.x >> 5;
  const int lane = threadIdx.x & 31;
  const long long row = (long long)blockIdx.x * 8 + wid;
  __bf16* p = P + row * Scol;

  float m = -3.0e38f;
  for (int i = lane; i < Scol; i += 32) m = fmaxf(m, (float)p[i]);
  for (int off = 16; off > 0; off >>= 1) m = fmaxf(m, __shfl_xor(m, off, 32));

  float s = 0.0f;
  for (int i = lane; i < Scol; i += 32) s += __expf((float)p[i] - m);
  for (int off = 16; off > 0; off >>= 1) s += __shfl_xor(s, off, 32);

  const float inv = 1.0f / s;
  for (int i = lane; i < Scol; i += 32)
    p[i] = (__bf16)(__expf((float)p[i] - m) * inv);
}

extern "C" void kernel_launch(void* const* d_in, const int* in_sizes, int n_in,
                              void* d_out, int out_size, void* d_ws, size_t ws_size,
                              hipStream_t stream) {
  constexpr int B = 4, S = 1024, D = 768, H = 12;
  constexpr long long BSD  = (long long)B * S * D;       // 3,145,728
  constexpr long long HDD  = (long long)H * D * D;       // 7,077,888
  constexpr long long BHSD = (long long)B * H * S * D;   // 37,748,736
  constexpr long long BHSS = (long long)B * H * S * S;   // 50,331,648
  constexpr long long HD   = (long long)H * D;           // 9216

  const float* x  = (const float*)d_in[0];
  const float* Wq = (const float*)d_in[1];
  const float* Wk = (const float*)d_in[2];
  const float* Wv = (const float*)d_in[3];
  const float* bq = (const float*)d_in[4];
  const float* bk = (const float*)d_in[5];
  const float* bv = (const float*)d_in[6];
  const float* Wo = (const float*)d_in[7];
  const float* bo = (const float*)d_in[8];

  char* ws = (char*)d_ws;
  size_t off = 0;
  auto take = [&](long long elems) -> __bf16* {
    __bf16* p = (__bf16*)(ws + off);
    off += ((size_t)elems * 2 + 255) & ~(size_t)255;
    return p;
  };
  __bf16* xb   = take(BSD);
  __bf16* wqbT = take(HDD);   // [h][e][d]  (W^T, [N][K])
  __bf16* wkbT = take(HDD);
  __bf16* wvbT = take(HDD);
  __bf16* wobT = take(HDD);   // [768][9216]
  __bf16* qb   = take(BHSD);  // [b,h][s][e]
  __bf16* kb   = take(BHSD);  // [b,h][t][e]  (== B^T for scores)
  __bf16* vbT  = take(BHSD);  // [b,h][e][t]  (written transposed)
  __bf16* pb   = take(BHSS);  // [b,h][s][t]
  __bf16* hb   = take(BHSD);  // [b][s][h*768+e] (concat-ready)

  // 1) fp32 -> bf16 (x plain; weights transposed to [N][K])
  mha_cvt_f32_bf16<<<4096, 256, 0, stream>>>(x, xb, BSD);
  {
    dim3 gT(D / 32, D / 32, H);
    mha_cvt_t_f32_bf16<<<gT, 256, 0, stream>>>(Wq, wqbT, D, D,
                                               (long long)D * D, (long long)D * D);
    mha_cvt_t_f32_bf16<<<gT, 256, 0, stream>>>(Wk, wkbT, D, D,
                                               (long long)D * D, (long long)D * D);
    mha_cvt_t_f32_bf16<<<gT, 256, 0, stream>>>(Wv, wvbT, D, D,
                                               (long long)D * D, (long long)D * D);
    dim3 gTo((int)HD / 32, D / 32, 1);
    mha_cvt_t_f32_bf16<<<gTo, 256, 0, stream>>>(Wo, wobT, (int)HD, D, 0, 0);
  }

  // 2) q/k = x[b] @ W[h] + bias[h]; v written transposed (v^T[e][t])
  {
    dim3 grid(S / 128, D / 128, B * H);
    mha_gemm_wmma_bf16<<<grid, 256, 0, stream>>>(
        xb, wqbT, qb, bq, D, D, D, D, H,
        (long long)S * D, 0, 0, (long long)D * D,
        (long long)H * S * D, (long long)S * D, D,
        1.0f, 0, /*outT=*/0, 0);
    mha_gemm_wmma_bf16<<<grid, 256, 0, stream>>>(
        xb, wkbT, kb, bk, D, D, D, D, H,
        (long long)S * D, 0, 0, (long long)D * D,
        (long long)H * S * D, (long long)S * D, D,
        1.0f, 0, /*outT=*/0, 0);
    mha_gemm_wmma_bf16<<<grid, 256, 0, stream>>>(
        xb, wvbT, vbT, bv, D, D, D, /*ldc=*/S, H,
        (long long)S * D, 0, 0, (long long)D * D,
        (long long)H * D * S, (long long)D * S, D,
        1.0f, 0, /*outT=*/1, 0);
  }

  // 3) scores[z] = (q[z] @ k[z]^T) / sqrt(D), causal  (M=N=1024, K=768)
  {
    dim3 grid(S / 128, S / 128, B * H);
    mha_gemm_wmma_bf16<<<grid, 256, 0, stream>>>(
        qb, kb, pb, nullptr, D, D, D, S, H,
        (long long)H * S * D, (long long)S * D,
        (long long)H * S * D, (long long)S * D,
        (long long)H * S * S, (long long)S * S,
        0, 1.0f / sqrtf((float)D), /*causal=*/1, 0, 0);
  }

  // 4) softmax in place: 48*1024 rows, 8 rows/block
  mha_softmax_rows<<<(B * H * S) / 8, 256, 0, stream>>>(pb, S);

  // 5) heads[z] = P[z] @ v[z]  (B operand is v^T); stored as [b][s][h*768+e]
  {
    dim3 grid(S / 128, D / 128, B * H);
    mha_gemm_wmma_bf16<<<grid, 256, 0, stream>>>(
        pb, vbT, hb, nullptr, S, S, /*ldb=*/S, (int)HD, H,
        (long long)H * S * S, (long long)S * S,
        (long long)H * D * S, (long long)D * S,
        (long long)S * HD, D,
        0, 1.0f, 0, 0, 0);
  }

  // 6) out = concat @ Wo + bo  (M=B*S=4096, N=768, K=9216, f32 output)
  {
    dim3 grid((B * S) / 128, D / 128, 1);
    mha_gemm_wmma_bf16<<<grid, 256, 0, stream>>>(
        hb, wobT, d_out, bo, (int)HD, (int)HD, (int)HD, D, 1,
        0, 0, 0, 0, 0, 0, 0,
        1.0f, 0, 0, /*outF32=*/1);
  }
}
